// LSTMWithStepwiseDropout_12962211299387
// MI455X (gfx1250) — matrix operand order repaired and checked by reading, not compiled, not run on hardware
//
#include <hip/hip_runtime.h>
#include <hip/hip_bf16.h>

// ---------------------------------------------------------------------------
// LSTM with stepwise dropout, MI455X (gfx1250, wave32, WMMA).
//
// Path A (workspace permitting):
//   1) pregate_gemm_kernel: P = x @ W_ih^T for ALL timesteps at once
//      (M=SEQ*B=32768, N=4096, K=1024) -> C/D-fragment-packed f32.
//   2) per-step kernel: acc seeded from P, serial K halves to H=1024
//      (h @ W_hh^T only), fused LSTM cell + stepwise dropout.
// Path B (small workspace): fused K=2048 ([x_t|h] @ [W_ih;W_hh]^T).
//
// Step kernel data movement (if toolchain exposes the gfx1250 async ops):
// double-buffered LDS staging via global_load_async_to_lds_b128 (ASYNCcnt),
// cooperative across the 8 waves of a block, WMMA operands then come from
// ds_load_b128 -> kills the 4x redundant B-fragment / 2x A-fragment L2
// traffic and overlaps L2 latency one K-stage ahead.
//
// All matrix math: v_wmma_f32_16x16x32_bf16 (bf16 in, f32 accumulate).
// Weights pre-packed into WMMA-B-fragment-native 1KB blocks (L2-resident).
// ---------------------------------------------------------------------------

#define SEQ    512
#define BATCH  64
#define IDIM   1024
#define HDIM   1024
#define KTOT   2048            // IDIM + HDIM fused K
#define NG     4096            // 4 * HDIM gate columns (i,f,g,o)
#define NTILES 256             // NG / 16
#define PDROP  0.1f

#if defined(__has_builtin)
#if __has_builtin(__builtin_amdgcn_global_load_async_to_lds_b128)
#define HAVE_ASYNC_LDS 1
#endif
#endif
#ifndef HAVE_ASYNC_LDS
#define HAVE_ASYNC_LDS 0
#endif

typedef __attribute__((ext_vector_type(16))) __bf16 v16bf;
typedef __attribute__((ext_vector_type(8)))  float  v8f;
typedef __attribute__((ext_vector_type(4)))  int    v4i;

#if HAVE_ASYNC_LDS
typedef __attribute__((address_space(1))) void as1_void;
typedef __attribute__((address_space(3))) void as3_void;
typedef __attribute__((address_space(1))) v4i  as1_v4i;
typedef __attribute__((address_space(3))) v4i  as3_v4i;
#endif

union Frag16 { v16bf v; uint4 q[2]; };
union FragC  { v8f v; float4 q[2]; };

__device__ __forceinline__ float sigmoidf_(float x) {
    return 1.0f / (1.0f + __expf(-x));
}

__device__ __forceinline__ void wait_async0() {
#if defined(__has_builtin)
#if __has_builtin(__builtin_amdgcn_s_wait_asynccnt)
    __builtin_amdgcn_s_wait_asynccnt(0);
#define DID_WAIT_ASYNC 1
#endif
#endif
#ifndef DID_WAIT_ASYNC
    asm volatile("s_wait_asynccnt 0x0" ::: "memory");
#endif
}

// ---------------------------------------------------------------------------
// Pack fused weights [W_ih ; W_hh] (fp32, row-major [n][k]) into bf16 WMMA
// B-fragment order. Fragment (ntile, kc) covers N = ntile*16..+16,
// K = kc*32..+32 and is 512 bf16 (1 KB): lane L holds column n0+L%16,
// K half selected by L>=16, as two contiguous uint4 loads.
// kc 0..31 -> W_ih (K = x part), kc 32..63 -> W_hh (K = h part).
// ---------------------------------------------------------------------------
__global__ __launch_bounds__(256) void pack_weights_kernel(
    const float* __restrict__ Wih, const float* __restrict__ Whh,
    __bf16* __restrict__ Wpack)
{
    long long e = (long long)blockIdx.x * blockDim.x + threadIdx.x;
    const long long total = (long long)NG * KTOT;
    if (e >= total) return;
    int within   = (int)(e & 511);
    long long fr = e >> 9;
    int L  = within >> 4;
    int j  = within & 15;
    int kc = (int)(fr & 63);
    int ntile = (int)(fr >> 6);
    int n = ntile * 16 + (L & 15);
    int k = kc * 32 + ((L >= 16) ? 8 : 0) + (j & 7) + ((j >= 8) ? 16 : 0);
    float v = (k < IDIM) ? Wih[(long long)n * IDIM + k]
                         : Whh[(long long)n * HDIM + (k - IDIM)];
    Wpack[e] = (__bf16)v;
}

// fp32 -> bf16 conversion of x, 4 elements per thread
__global__ __launch_bounds__(256) void convert_x_kernel(
    const float* __restrict__ x, __bf16* __restrict__ xb, long long n4)
{
    long long i = (long long)blockIdx.x * blockDim.x + threadIdx.x;
    if (i >= n4) return;
    float4 v = ((const float4*)x)[i];
    union { __bf16 h[4]; unsigned long long u; } o;
    o.h[0] = (__bf16)v.x; o.h[1] = (__bf16)v.y;
    o.h[2] = (__bf16)v.z; o.h[3] = (__bf16)v.w;
    ((unsigned long long*)xb)[i] = o.u;
}

// zero h0/h1/c, fuse biases
__global__ __launch_bounds__(256) void init_state_kernel(
    __bf16* __restrict__ h0, __bf16* __restrict__ h1, float* __restrict__ c,
    float* __restrict__ bias, const float* __restrict__ b_ih,
    const float* __restrict__ b_hh)
{
    int i = blockIdx.x * blockDim.x + threadIdx.x;
    if (i < BATCH * HDIM) {
        h0[i] = (__bf16)0.0f;
        h1[i] = (__bf16)0.0f;
        c[i]  = 0.0f;
    }
    if (i < NG) bias[i] = b_ih[i] + b_hh[i];
}

// ---------------------------------------------------------------------------
// Pregates: P = x @ W_ih^T for all SEQ*B rows in one shot.
// Each wave: one M-tile (16 rows) x 4 N-tiles (64 cols), K = 1024.
// Output stored C/D-fragment-packed f32: [mtile][ntile][lane][8] so the
// step kernel seeds its accumulators with two uint4 loads per lane.
// Grid: 2048 mtiles * 64 ngroups = 131072 waves = 16384 blocks.
// ---------------------------------------------------------------------------
__global__ __launch_bounds__(256) void pregate_gemm_kernel(
    const __bf16* __restrict__ xb, const __bf16* __restrict__ Wpack,
    float* __restrict__ Ppack)
{
    const int tid  = threadIdx.x;
    const int lane = tid & 31;
    const int wave = tid >> 5;
    const int wid  = blockIdx.x * 8 + wave;   // 0..131071
    const int mtile  = wid >> 6;              // 0..2047
    const int ngroup = wid & 63;              // ntiles ngroup*4 .. +4

    const int row   = mtile * 16 + (lane & 15);
    const int koff  = (lane >> 4) << 3;
    const int laneB = lane << 4;
    const __bf16* Arow = xb + (long long)row * IDIM;

    v8f acc[4] = {v8f{}, v8f{}, v8f{}, v8f{}};
    const __bf16* Bb[4];
    #pragma unroll
    for (int q = 0; q < 4; ++q)
        Bb[q] = Wpack + ((long long)(ngroup * 4 + q) * 64) * 512 + laneB;

    #pragma unroll 2
    for (int kc = 0; kc < 32; ++kc) {         // K = IDIM only (W_ih half)
        Frag16 A;
        const __bf16* ap = Arow + kc * 32;
        A.q[0] = *(const uint4*)(ap + koff);
        A.q[1] = *(const uint4*)(ap + koff + 16);
        #pragma unroll
        for (int q = 0; q < 4; ++q) {
            Frag16 B;
            const __bf16* bp = Bb[q] + (long long)kc * 512;
            B.q[0] = ((const uint4*)bp)[0];
            B.q[1] = ((const uint4*)bp)[1];
            acc[q] = __builtin_amdgcn_wmma_f32_16x16x32_bf16(
                         false, A.v, false, B.v, (short)0, acc[q],
                         false, false);
        }
    }

    #pragma unroll
    for (int q = 0; q < 4; ++q) {
        FragC out; out.v = acc[q];
        float* dst = Ppack
            + ((size_t)(mtile * NTILES + (ngroup * 4 + q)) * 256)
            + lane * 8;
        ((float4*)dst)[0] = out.q[0];
        ((float4*)dst)[1] = out.q[1];
    }
}

#if HAVE_ASYNC_LDS
// Cast helpers: C-style addrspacecast (generic -> AS) then bitcast in-AS.
__device__ __forceinline__ as1_v4i* to_as1(const void* p) {
    return (as1_v4i*)(as1_void*)(void*)p;
}
__device__ __forceinline__ as3_v4i* to_as3(void* p) {
    return (as3_v4i*)(as3_void*)p;
}

// Issue the cooperative async stage for K-chunk kc into Abuf/Bbuf.
// 8 waves x 2 async b128 ops: wave w stages half (w&1) of fragment (w>>1)
// of both the A side (batch-tile f) and B side (gate f, ntile f*64+jb).
// All 1KB fragments land in LDS in the exact register-fragment layout.
__device__ __forceinline__ void stage_async(
    int kc, int jb, int wave, int lane,
    const __bf16* __restrict__ xt, const __bf16* __restrict__ hin,
    const __bf16* __restrict__ Wpack,
    __bf16* Abuf, __bf16* Bbuf)
{
    const int f  = wave >> 1;
    const int hh = wave & 1;
    // B-matrix fragment: identity copy of the packed 1KB block
    const __bf16* bsrc = Wpack
        + ((size_t)((f * 64 + jb) * 64 + kc)) * 512 + hh * 256 + lane * 8;
    __bf16* bdst = Bbuf + f * 512 + hh * 256 + lane * 8;
    __builtin_amdgcn_global_load_async_to_lds_b128(
        to_as1(bsrc), to_as3(bdst), 0, 0);
    // A-matrix fragment: gather 16B units into register-fragment order
    const int u    = hh * 32 + lane;      // 16B unit 0..63 of the fragment
    const int Lp   = u >> 1;              // consuming lane
    const int cch  = u & 1;               // which of its two 16B chunks
    const int arow = f * 16 + (Lp & 15);
    const int kk   = ((Lp >= 16) ? 8 : 0) + cch * 16;
    const __bf16* asrc = (kc < 32)
        ? (xt  + (size_t)arow * IDIM + kc * 32 + kk)
        : (hin + (size_t)arow * HDIM + (kc - 32) * 32 + kk);
    __bf16* adst = Abuf + f * 512 + u * 8;
    __builtin_amdgcn_global_load_async_to_lds_b128(
        to_as1(asrc), to_as3(adst), 0, 0);
}
#endif

// ---------------------------------------------------------------------------
// One recurrent step, fully fused. Block jb (0..63) owns h-columns
// jb*16..+16 and computes all four gate tiles (i,f,g,o) for all 64 batch
// rows; wave w: mt = w>>1 (batch rows), gates {0,1} or {2,3} (w&1).
// use_pregate: seed acc from Ppack, loop kc 32..63 (h only);
// else: acc from 0, loop kc 0..63 ([x_t | h]).
// Gates exchanged across waves via LDS, then LSTM cell + dropout.
// ---------------------------------------------------------------------------
__global__ __launch_bounds__(256) void lstm_step_kernel(
    const __bf16* __restrict__ xt,      // [BATCH][IDIM] bf16, step t
    const __bf16* __restrict__ hin,     // [BATCH][HDIM] bf16
    __bf16*       __restrict__ hout,    // [BATCH][HDIM] bf16
    float*        __restrict__ c,       // [BATCH][HDIM] f32 (in place)
    const __bf16* __restrict__ Wpack,   // packed fragments
    const float*  __restrict__ Ppack,   // packed pregates (or null)
    const float*  __restrict__ bias,    // [NG]
    const float*  __restrict__ mask_t,  // keep_mask + t*BATCH*HDIM
    const int*    __restrict__ training,
    float*        __restrict__ out,     // [BATCH][HDIM] f32 (final h)
    int t, int use_pregate, int write_out)
{
    __shared__ float Gs[4][4][16][16];   // [mt][gate][m][n] = 16 KB

    const int tid  = threadIdx.x;
    const int lane = tid & 31;
    const int wave = tid >> 5;
    const int mt   = wave >> 1;          // batch tile 0..3
    const int gp   = wave & 1;           // gate pair: {0,1} or {2,3}
    const int jb   = blockIdx.x;         // h-column tile 0..63
    const int m0   = mt * 16;
    const int row  = m0 + (lane & 15);
    const int koff = (lane >> 4) << 3;   // A frag: 0 or 8 within K chunk
    const int laneB = lane << 4;         // B frag: lane*16 elements

    const int nt0 = (2 * gp) * 64 + jb;  // gate 2*gp,   ntile
    const int nt1 = nt0 + 64;            // gate 2*gp+1, ntile

    v8f acc0 = {};
    v8f acc1 = {};
    const int kcbeg = use_pregate ? 32 : 0;
    if (use_pregate) {
        const int mtile_g = t * 4 + mt;
        const float* p0 = Ppack + ((size_t)(mtile_g * NTILES + nt0) * 256)
                                + lane * 8;
        const float* p1 = Ppack + ((size_t)(mtile_g * NTILES + nt1) * 256)
                                + lane * 8;
        FragC c0, c1;
        c0.q[0] = ((const float4*)p0)[0]; c0.q[1] = ((const float4*)p0)[1];
        c1.q[0] = ((const float4*)p1)[0]; c1.q[1] = ((const float4*)p1)[1];
        acc0 = c0.v;
        acc1 = c1.v;
    }

#if HAVE_ASYNC_LDS
    // Double-buffered cooperative LDS staging via ASYNCcnt DMA.
    __shared__ __align__(32) __bf16 Asm[2][4 * 512];   // 4 A frags / stage
    __shared__ __align__(32) __bf16 Bsm[2][4 * 512];   // 4 B frags / stage

    stage_async(kcbeg, jb, wave, lane, xt, hin, Wpack, Asm[0], Bsm[0]);
    wait_async0();
    __syncthreads();

    const int g0lds = 2 * gp;
    for (int kc = kcbeg; kc < 64; ++kc) {
        const int cur = (kc - kcbeg) & 1;
        const int nxt = cur ^ 1;
        if (kc + 1 < 64)
            stage_async(kc + 1, jb, wave, lane, xt, hin, Wpack,
                        Asm[nxt], Bsm[nxt]);
        Frag16 A, B0, B1;
        const __bf16* ap = Asm[cur] + mt * 512 + lane * 16;
        const __bf16* b0 = Bsm[cur] + g0lds * 512 + lane * 16;
        const __bf16* b1 = b0 + 512;
        A.q[0]  = ((const uint4*)ap)[0];  A.q[1]  = ((const uint4*)ap)[1];
        B0.q[0] = ((const uint4*)b0)[0];  B0.q[1] = ((const uint4*)b0)[1];
        B1.q[0] = ((const uint4*)b1)[0];  B1.q[1] = ((const uint4*)b1)[1];
        acc0 = __builtin_amdgcn_wmma_f32_16x16x32_bf16(
                   false, A.v, false, B0.v, (short)0, acc0, false, false);
        acc1 = __builtin_amdgcn_wmma_f32_16x16x32_bf16(
                   false, A.v, false, B1.v, (short)0, acc1, false, false);
        if (kc + 1 < 64) {
            wait_async0();       // own stage transfers done
            __syncthreads();     // everyone's transfers done
        }
    }
#else
    // Fallback: direct global fragment loads (weights are L2-resident).
    const __bf16* Arow_x = xt  + (long long)row * IDIM;
    const __bf16* Arow_h = hin + (long long)row * HDIM;
    const __bf16* Bb0 = Wpack + ((long long)nt0 * 64) * 512 + laneB;
    const __bf16* Bb1 = Wpack + ((long long)nt1 * 64) * 512 + laneB;

    #pragma unroll 4
    for (int kc = kcbeg; kc < 64; ++kc) {
        const __bf16* ap = (kc < 32) ? (Arow_x + kc * 32)
                                     : (Arow_h + (kc - 32) * 32);
        Frag16 A, B0, B1;
        A.q[0] = *(const uint4*)(ap + koff);
        A.q[1] = *(const uint4*)(ap + koff + 16);
        const __bf16* b0 = Bb0 + (long long)kc * 512;
        const __bf16* b1 = Bb1 + (long long)kc * 512;
        B0.q[0] = ((const uint4*)b0)[0];  B0.q[1] = ((const uint4*)b0)[1];
        B1.q[0] = ((const uint4*)b1)[0];  B1.q[1] = ((const uint4*)b1)[1];
        if (kc + 1 < 64) {
            __builtin_prefetch(b0 + 512, 0, 1);
            __builtin_prefetch(b1 + 512, 0, 1);
        }
        acc0 = __builtin_amdgcn_wmma_f32_16x16x32_bf16(
                   false, A.v, false, B0.v, (short)0, acc0, false, false);
        acc1 = __builtin_amdgcn_wmma_f32_16x16x32_bf16(
                   false, A.v, false, B1.v, (short)0, acc1, false, false);
    }
#endif

    // C/D layout: vgpr r, lanes 0-15: (m=r, n=lane); lanes 16-31: (m=8+r).
    const int n     = lane & 15;
    const int mhalf = (lane >> 4) << 3;
    const int g0    = 2 * gp;
    const int nc0   = g0 * HDIM + jb * 16 + n;
    const int nc1   = nc0 + HDIM;
    const float bi0 = bias[nc0];
    const float bi1 = bias[nc1];
    #pragma unroll
    for (int r = 0; r < 8; ++r) {
        int mm = mhalf + r;
        Gs[mt][g0    ][mm][n] = acc0[r] + bi0;
        Gs[mt][g0 + 1][mm][n] = acc1[r] + bi1;
    }
    __syncthreads();

    const bool  train = (*training != 0);
    const float scale = 1.0f / (1.0f - PDROP);
    for (int e = tid; e < BATCH * 16; e += 256) {
        int b  = e >> 4;
        int j  = e & 15;
        int bt = b >> 4;
        int bm = b & 15;
        float ig = sigmoidf_(Gs[bt][0][bm][j]);
        float fg = sigmoidf_(Gs[bt][1][bm][j]);
        float gg = tanhf    (Gs[bt][2][bm][j]);
        float og = sigmoidf_(Gs[bt][3][bm][j]);
        long long idx = (long long)b * HDIM + (jb * 16 + j);
        float cc = fg * c[idx] + ig * gg;
        c[idx] = cc;
        float hh = og * tanhf(cc);
        if (train) hh *= mask_t[idx] * scale;
        hout[idx] = (__bf16)hh;
        if (write_out) out[idx] = hh;
    }
}

// ---------------------------------------------------------------------------
extern "C" void kernel_launch(void* const* d_in, const int* in_sizes, int n_in,
                              void* d_out, int out_size, void* d_ws,
                              size_t ws_size, hipStream_t stream)
{
    (void)in_sizes; (void)n_in; (void)out_size;
    const float* x         = (const float*)d_in[0];
    const float* keep_mask = (const float*)d_in[1];
    const float* W_ih      = (const float*)d_in[2];
    const float* W_hh      = (const float*)d_in[3];
    const float* b_ih      = (const float*)d_in[4];
    const float* b_hh      = (const float*)d_in[5];
    const int*   training  = (const int*)d_in[6];
    float*       out       = (float*)d_out;

    // workspace layout
    char* ws = (char*)d_ws;
    size_t off = 0;
    __bf16* Wpack = (__bf16*)(ws + off); off += (size_t)NG * KTOT * 2;   // 16 MB
    float*  bias  = (float*)(ws + off);  off += (size_t)NG * 4;          // 16 KB
    __bf16* h0    = (__bf16*)(ws + off); off += (size_t)BATCH * HDIM * 2;
    __bf16* h1    = (__bf16*)(ws + off); off += (size_t)BATCH * HDIM * 2;
    float*  c     = (float*)(ws + off);  off += (size_t)BATCH * HDIM * 4;
    off = (off + 255) & ~(size_t)255;
    __bf16* xb    = (__bf16*)(ws + off); off += (size_t)SEQ * BATCH * IDIM * 2; // 64 MB
    off = (off + 255) & ~(size_t)255;
    float*  Ppack = (float*)(ws + off);
    size_t need_full = off + (size_t)SEQ * BATCH * NG * 4;               // +512 MB

    // deterministic path choice (ws_size is fixed across calls)
    const int use_pre = (ws_size >= need_full) ? 1 : 0;

    {
        long long total = (long long)NG * KTOT;
        int blocks = (int)((total + 255) / 256);
        pack_weights_kernel<<<blocks, 256, 0, stream>>>(W_ih, W_hh, Wpack);
    }
    {
        long long n4 = (long long)SEQ * BATCH * IDIM / 4;
        int blocks = (int)((n4 + 255) / 256);
        convert_x_kernel<<<blocks, 256, 0, stream>>>(x, xb, n4);
    }
    init_state_kernel<<<(BATCH * HDIM + 255) / 256, 256, 0, stream>>>(
        h0, h1, c, bias, b_ih, b_hh);

    if (use_pre) {
        // 2048 mtiles * 64 ngroups, 8 waves/block -> 16384 blocks
        pregate_gemm_kernel<<<16384, 256, 0, stream>>>(xb, Wpack, Ppack);
    }

    for (int t = 0; t < SEQ; ++t) {
        const __bf16* xt   = xb + (long long)t * BATCH * IDIM;
        const __bf16* hin  = (t & 1) ? h1 : h0;
        __bf16*       hout = (t & 1) ? h0 : h1;
        const float*  mt   = keep_mask + (long long)t * BATCH * HDIM;
        lstm_step_kernel<<<64, 256, 0, stream>>>(
            xt, hin, hout, c, Wpack, use_pre ? Ppack : (const float*)nullptr,
            bias, mt, training, out, t, use_pre, (t == SEQ - 1) ? 1 : 0);
    }
}
